// MistralAttention_streaming_47923245089255
// MI455X (gfx1250) — matrix operand order repaired and checked by reading, not compiled, not run on hardware
//
#include <hip/hip_runtime.h>
#include <stddef.h>

// ---------------------------------------------------------------------------
// CDNA5 (gfx1250) Mistral attention forward.
// bf16 WMMA (v_wmma_f32_16x16x32_bf16) for all GEMMs, fp32 accumulate,
// flash-attention streaming softmax, double-buffered LDS GEMM main loop,
// global_prefetch for upcoming tiles.
// ---------------------------------------------------------------------------

typedef __attribute__((ext_vector_type(16))) __bf16  bf16x16;
typedef __attribute__((ext_vector_type(8)))  float   f32x8;
typedef __attribute__((ext_vector_type(4)))  unsigned int u32x4;

union Frag {
    bf16x16 v;
    u32x4   q[2];
};

__device__ __forceinline__ unsigned short f2bf(float f) {
    union { float f; unsigned u; } x; x.f = f;
    unsigned u = x.u;
    unsigned r = (u + 0x7FFFu + ((u >> 16) & 1u)) >> 16;   // RNE
    return (unsigned short)r;
}
__device__ __forceinline__ float bf2f(unsigned short b) {
    union { unsigned u; float f; } x; x.u = ((unsigned)b) << 16;
    return x.f;
}

// ---------------------------------------------------------------------------
// fp32 -> bf16 cast (grid-stride)
// ---------------------------------------------------------------------------
__global__ void mistral_cast_bf16(const float* __restrict__ in,
                                  unsigned short* __restrict__ out, size_t n) {
    size_t i = (size_t)blockIdx.x * blockDim.x + threadIdx.x;
    size_t stride = (size_t)gridDim.x * blockDim.x;
    for (; i < n; i += stride) out[i] = f2bf(in[i]);
}

// ---------------------------------------------------------------------------
// RoPE on bf16 [S, nh*128], half-rotation as in reference.
// ---------------------------------------------------------------------------
__global__ void mistral_rope(unsigned short* __restrict__ X,
                             const int* __restrict__ pos, int nh, int S) {
    int idx = blockIdx.x * blockDim.x + threadIdx.x;
    int total = S * nh * 64;
    if (idx >= total) return;
    int d  = idx & 63;
    int hh = (idx >> 6) % nh;
    int s  = idx / (64 * nh);
    float p   = (float)pos[s];
    float inv = __expf(-0.14391157f * (float)d);   // 10000^(-2d/128)
    float ang = p * inv;
    float sn, cs;
    __sincosf(ang, &sn, &cs);
    size_t base = (size_t)s * ((size_t)nh * 128) + (size_t)hh * 128 + d;
    float x1 = bf2f(X[base]);
    float x2 = bf2f(X[base + 64]);
    X[base]      = f2bf(x1 * cs - x2 * sn);
    X[base + 64] = f2bf(x2 * cs + x1 * sn);
}

// ---------------------------------------------------------------------------
// bf16 GEMM: C[M,N] = A[M,K] * B[K,N], fp32 accum, compile-time shapes.
// BM=BN=128, BK=32, 256 threads = 8 waves, wave owns 16x128 strip.
// Register-staged double-buffered LDS; prefetch of tile k+2.
// ---------------------------------------------------------------------------
template <int M, int N, int K, bool OUTF32>
__global__ __launch_bounds__(256)
void mistral_gemm_bf16(const unsigned short* __restrict__ A,
                       const unsigned short* __restrict__ B,
                       void* __restrict__ C) {
    __shared__ unsigned short ldsA[2][128 * 40];   // [m][k], row stride 40
    __shared__ unsigned short ldsBt[2][128 * 40];  // [n][k], row stride 40

    const int bn = blockIdx.x * 128;
    const int bm = blockIdx.y * 128;
    const int tid  = threadIdx.x;
    const int wv   = tid >> 5;
    const int lane = tid & 31;
    const int lrow = lane & 15;
    const int hlf  = lane >> 4;
    const int mBase = wv * 16;

    // per-thread tile-load mapping (2 chunks of 8 bf16 for A and for B)
    int aRow[2], aKc[2], bKr[2], bNc[2];
#pragma unroll
    for (int c = 0; c < 2; ++c) {
        int id = tid + c * 256;
        aRow[c] = id >> 2;  aKc[c] = (id & 3) * 8;
        bKr[c] = id >> 4;   bNc[c] = (id & 15) * 8;
    }

    f32x8 acc[8];
#pragma unroll
    for (int t = 0; t < 8; ++t) acc[t] = (f32x8){};

    const int NK = K / 32;
    u32x4 ra[2], rb[2];

    // stage tile 0
#pragma unroll
    for (int c = 0; c < 2; ++c) {
        ra[c] = *(const u32x4*)&A[(size_t)(bm + aRow[c]) * K + aKc[c]];
        rb[c] = *(const u32x4*)&B[(size_t)bKr[c] * N + bn + bNc[c]];
    }
#pragma unroll
    for (int c = 0; c < 2; ++c) {
        *(u32x4*)&ldsA[0][aRow[c] * 40 + aKc[c]] = ra[c];
        const unsigned short* p = (const unsigned short*)&rb[c];
#pragma unroll
        for (int i = 0; i < 8; ++i)
            ldsBt[0][(bNc[c] + i) * 40 + bKr[c]] = p[i];
    }
    __syncthreads();

    for (int kt = 0; kt < NK; ++kt) {
        const int cur = kt & 1;
        const int nxt = cur ^ 1;

        // prefetch tile kt+2 into L2/WGP$
        if (kt + 2 < NK) {
            int kp = (kt + 2) * 32;
            __builtin_prefetch(&A[(size_t)(bm + aRow[0]) * K + kp + aKc[0]], 0, 1);
            __builtin_prefetch(&B[(size_t)(kp + bKr[0]) * N + bn + bNc[0]], 0, 1);
        }

        // stage tile kt+1 into registers (overlaps with WMMA below)
        if (kt + 1 < NK) {
            int kn = (kt + 1) * 32;
#pragma unroll
            for (int c = 0; c < 2; ++c) {
                ra[c] = *(const u32x4*)&A[(size_t)(bm + aRow[c]) * K + kn + aKc[c]];
                rb[c] = *(const u32x4*)&B[(size_t)(kn + bKr[c]) * N + bn + bNc[c]];
            }
        }

        // compute on buffer `cur`
        Frag a;
        int ro = (mBase + lrow) * 40;
        a.q[0] = *(const u32x4*)&ldsA[cur][ro + hlf * 8];
        a.q[1] = *(const u32x4*)&ldsA[cur][ro + 16 + hlf * 8];
#pragma unroll
        for (int t = 0; t < 8; ++t) {
            Frag b;
            int rbo = (t * 16 + lrow) * 40;
            b.q[0] = *(const u32x4*)&ldsBt[cur][rbo + hlf * 8];
            b.q[1] = *(const u32x4*)&ldsBt[cur][rbo + 16 + hlf * 8];
            acc[t] = __builtin_amdgcn_wmma_f32_16x16x32_bf16(
                false, a.v, false, b.v, (short)0, acc[t], false, false);
        }

        // write tile kt+1 into the other LDS buffer
        if (kt + 1 < NK) {
#pragma unroll
            for (int c = 0; c < 2; ++c) {
                *(u32x4*)&ldsA[nxt][aRow[c] * 40 + aKc[c]] = ra[c];
                const unsigned short* p = (const unsigned short*)&rb[c];
#pragma unroll
                for (int i = 0; i < 8; ++i)
                    ldsBt[nxt][(bNc[c] + i) * 40 + bKr[c]] = p[i];
            }
        }
        __syncthreads();
    }

#pragma unroll
    for (int t = 0; t < 8; ++t) {
#pragma unroll
        for (int r = 0; r < 8; ++r) {
            int row = bm + mBase + hlf * 8 + r;
            int col = bn + t * 16 + lrow;
            if constexpr (OUTF32)
                ((float*)C)[(size_t)row * N + col] = acc[t][r];
            else
                ((unsigned short*)C)[(size_t)row * N + col] = f2bf(acc[t][r]);
        }
    }
}

// ---------------------------------------------------------------------------
// Flash attention, causal (window >= S).
// Block = (head h, 128 query rows). 8 waves, wave owns 16 q rows.
// KV streamed in 64-row tiles. GQA: kv head = h/4.
// ---------------------------------------------------------------------------
__global__ __launch_bounds__(256)
void mistral_flash_attn(const unsigned short* __restrict__ Qb,
                        const unsigned short* __restrict__ Kb,
                        const unsigned short* __restrict__ Vb,
                        unsigned short* __restrict__ Ob) {
    __shared__ unsigned short ldsQ[128 * 136];      // [qrow][d], stride 136
    __shared__ unsigned short ldsK[64 * 136];       // [kvrow][d]
    __shared__ unsigned short ldsVt[128 * 72];      // [d][kvrow], stride 72
    __shared__ unsigned short ldsP[8 * 16 * 72];    // per-wave P staging

    const int h     = blockIdx.x;
    const int qBase = blockIdx.y * 128;
    const int kvh   = h >> 2;
    const int tid   = threadIdx.x;
    const int wv    = tid >> 5;
    const int lane  = tid & 31;
    const int lrow  = lane & 15;
    const int hlf   = lane >> 4;
    const int mBase = wv * 16;
    const float scale = 0.08838834764831845f;       // 1/sqrt(128)

    // load Q tile [128][128]
#pragma unroll
    for (int c = 0; c < 8; ++c) {
        int id  = tid + c * 256;
        int row = id >> 4;
        int kc  = id & 15;
        *(u32x4*)&ldsQ[row * 136 + kc * 8] =
            *(const u32x4*)&Qb[(size_t)(qBase + row) * 4096 + h * 128 + kc * 8];
    }
    __syncthreads();

    // preload Q A-frags for all 4 K-steps (head dim 128 = 4 x 32)
    Frag qa[4];
#pragma unroll
    for (int ks = 0; ks < 4; ++ks) {
        int ro = (mBase + lrow) * 136 + ks * 32;
        qa[ks].q[0] = *(const u32x4*)&ldsQ[ro + hlf * 8];
        qa[ks].q[1] = *(const u32x4*)&ldsQ[ro + 16 + hlf * 8];
    }

    f32x8 o[8];
    float mprev[8], l[8];
#pragma unroll
    for (int t = 0; t < 8; ++t) o[t] = (f32x8){};
#pragma unroll
    for (int r = 0; r < 8; ++r) { mprev[r] = -3.0e38f; l[r] = 0.f; }

    const int nkv = (qBase + 128) / 64;
    for (int jb = 0; jb < nkv; ++jb) {
        const int jBase = jb * 64;
        __syncthreads();   // previous iteration's frag reads done

        // prefetch next KV tile
        if (jb + 1 < nkv) {
            int jn = jBase + 64;
            int row = tid >> 4, kc = tid & 15;
            __builtin_prefetch(&Kb[(size_t)(jn + row) * 1024 + kvh * 128 + kc * 8], 0, 1);
            __builtin_prefetch(&Vb[(size_t)(jn + row) * 1024 + kvh * 128 + kc * 8], 0, 1);
        }

        // K tile [64][128]
#pragma unroll
        for (int c = 0; c < 4; ++c) {
            int id  = tid + c * 256;
            int row = id >> 4;
            int kc  = id & 15;
            *(u32x4*)&ldsK[row * 136 + kc * 8] =
                *(const u32x4*)&Kb[(size_t)(jBase + row) * 1024 + kvh * 128 + kc * 8];
        }
        // V tile transposed -> [d][kv]
#pragma unroll
        for (int c = 0; c < 4; ++c) {
            int id  = tid + c * 256;
            int row = id >> 4;
            int kc  = id & 15;
            u32x4 dv = *(const u32x4*)&Vb[(size_t)(jBase + row) * 1024 + kvh * 128 + kc * 8];
            const unsigned short* p = (const unsigned short*)&dv;
#pragma unroll
            for (int i = 0; i < 8; ++i)
                ldsVt[(kc * 8 + i) * 72 + row] = p[i];
        }
        __syncthreads();

        // S = Q * K^T   (4 n-tiles of kv, 4 k-steps of head dim)
        f32x8 sc[4];
#pragma unroll
        for (int t = 0; t < 4; ++t) sc[t] = (f32x8){};
#pragma unroll
        for (int t = 0; t < 4; ++t) {
#pragma unroll
            for (int ks = 0; ks < 4; ++ks) {
                Frag kb;
                int ro = (t * 16 + lrow) * 136 + ks * 32;
                kb.q[0] = *(const u32x4*)&ldsK[ro + hlf * 8];
                kb.q[1] = *(const u32x4*)&ldsK[ro + 16 + hlf * 8];
                sc[t] = __builtin_amdgcn_wmma_f32_16x16x32_bf16(
                    false, qa[ks].v, false, kb.v, (short)0, sc[t], false, false);
            }
        }

        // online softmax update (row = hlf*8+r, cols across 16-lane halves)
#pragma unroll
        for (int r = 0; r < 8; ++r) {
            int i = qBase + mBase + hlf * 8 + r;
            float mt = -3.0e38f;
#pragma unroll
            for (int t = 0; t < 4; ++t) {
                int j = jBase + t * 16 + lrow;
                float x = sc[t][r] * scale;
                if (j > i) x = -3.0e38f;           // causal mask
                sc[t][r] = x;
                mt = fmaxf(mt, x);
            }
#pragma unroll
            for (int off = 1; off < 16; off <<= 1)
                mt = fmaxf(mt, __shfl_xor(mt, off, 32));
            float mnew = fmaxf(mprev[r], mt);
            float alpha = __expf(mprev[r] - mnew);
            float rs = 0.f;
#pragma unroll
            for (int t = 0; t < 4; ++t) {
                float p = __expf(sc[t][r] - mnew);
                sc[t][r] = p;
                rs += p;
            }
#pragma unroll
            for (int off = 1; off < 16; off <<= 1)
                rs += __shfl_xor(rs, off, 32);
            l[r] = l[r] * alpha + rs;
#pragma unroll
            for (int t = 0; t < 8; ++t) o[t][r] *= alpha;
            mprev[r] = mnew;
        }

        // P: C-layout -> A-layout via wave-private LDS staging (bf16)
        unsigned short* pb = &ldsP[wv * 16 * 72];
#pragma unroll
        for (int t = 0; t < 4; ++t)
#pragma unroll
            for (int r = 0; r < 8; ++r)
                pb[(hlf * 8 + r) * 72 + t * 16 + lrow] = f2bf(sc[t][r]);

        Frag pa[2];
#pragma unroll
        for (int ks = 0; ks < 2; ++ks) {
            int ro = lrow * 72 + ks * 32;
            pa[ks].q[0] = *(const u32x4*)&pb[ro + hlf * 8];
            pa[ks].q[1] = *(const u32x4*)&pb[ro + 16 + hlf * 8];
        }

        // O += P * V   (8 n-tiles of head dim, 2 k-steps of kv)
#pragma unroll
        for (int t = 0; t < 8; ++t) {
#pragma unroll
            for (int ks = 0; ks < 2; ++ks) {
                Frag vb;
                int ro = (t * 16 + lrow) * 72 + ks * 32;
                vb.q[0] = *(const u32x4*)&ldsVt[ro + hlf * 8];
                vb.q[1] = *(const u32x4*)&ldsVt[ro + 16 + hlf * 8];
                o[t] = __builtin_amdgcn_wmma_f32_16x16x32_bf16(
                    false, pa[ks].v, false, vb.v, (short)0, o[t], false, false);
            }
        }
    }

    // epilogue: normalize, write bf16 [S, 4096]
#pragma unroll
    for (int t = 0; t < 8; ++t) {
#pragma unroll
        for (int r = 0; r < 8; ++r) {
            int row = qBase + mBase + hlf * 8 + r;
            int col = t * 16 + lrow;
            float inv = 1.0f / l[r];
            Ob[(size_t)row * 4096 + h * 128 + col] = f2bf(o[t][r] * inv);
        }
    }
}

// ---------------------------------------------------------------------------
// Launch
// ---------------------------------------------------------------------------
extern "C" void kernel_launch(void* const* d_in, const int* in_sizes, int n_in,
                              void* d_out, int out_size, void* d_ws, size_t ws_size,
                              hipStream_t stream) {
    const int S = 2048, H = 4096, NH = 32, NKV = 8;
    const int KVW = NKV * 128;   // 1024

    const float* hs  = (const float*)d_in[0];
    const int*   pos = (const int*)d_in[1];
    const float* qw  = (const float*)d_in[2];
    const float* kw  = (const float*)d_in[3];
    const float* vw  = (const float*)d_in[4];
    const float* ow  = (const float*)d_in[5];
    float* out = (float*)d_out;

    // workspace carve-up (bf16), ~143 MB total
    char* w = (char*)d_ws;
    unsigned short* Xb  = (unsigned short*)w; w += (size_t)S * H * 2;
    unsigned short* Wq  = (unsigned short*)w; w += (size_t)H * H * 2;
    unsigned short* Wk  = (unsigned short*)w; w += (size_t)H * KVW * 2;
    unsigned short* Wv  = (unsigned short*)w; w += (size_t)H * KVW * 2;
    unsigned short* Wo  = (unsigned short*)w; w += (size_t)H * H * 2;
    unsigned short* Qb  = (unsigned short*)w; w += (size_t)S * H * 2;
    unsigned short* Kbf = (unsigned short*)w; w += (size_t)S * KVW * 2;
    unsigned short* Vbf = (unsigned short*)w; w += (size_t)S * KVW * 2;
    unsigned short* Ab  = (unsigned short*)w; w += (size_t)S * H * 2;

    // 1) fp32 -> bf16
    mistral_cast_bf16<<<2048, 256, 0, stream>>>(hs, Xb, (size_t)S * H);
    mistral_cast_bf16<<<2048, 256, 0, stream>>>(qw, Wq, (size_t)H * H);
    mistral_cast_bf16<<<2048, 256, 0, stream>>>(kw, Wk, (size_t)H * KVW);
    mistral_cast_bf16<<<2048, 256, 0, stream>>>(vw, Wv, (size_t)H * KVW);
    mistral_cast_bf16<<<2048, 256, 0, stream>>>(ow, Wo, (size_t)H * H);

    // 2) QKV projections (bf16 out)
    mistral_gemm_bf16<2048, 4096, 4096, false>
        <<<dim3(H / 128, S / 128), 256, 0, stream>>>(Xb, Wq, Qb);
    mistral_gemm_bf16<2048, 1024, 4096, false>
        <<<dim3(KVW / 128, S / 128), 256, 0, stream>>>(Xb, Wk, Kbf);
    mistral_gemm_bf16<2048, 1024, 4096, false>
        <<<dim3(KVW / 128, S / 128), 256, 0, stream>>>(Xb, Wv, Vbf);

    // 3) RoPE on Q and K
    mistral_rope<<<(S * NH * 64 + 255) / 256, 256, 0, stream>>>(Qb, pos, NH, S);
    mistral_rope<<<(S * NKV * 64 + 255) / 256, 256, 0, stream>>>(Kbf, pos, NKV, S);

    // 4) flash attention -> Ab [S, 4096] bf16
    mistral_flash_attn<<<dim3(NH, S / 128), 256, 0, stream>>>(Qb, Kbf, Vbf, Ab);

    // 5) output projection -> fp32 d_out
    mistral_gemm_bf16<2048, 4096, 4096, true>
        <<<dim3(H / 128, S / 128), 256, 0, stream>>>(Ab, Wo, out);
}
